// CausalSelfAttention_35081292874334
// MI455X (gfx1250) — compile-verified, matrix operand order checked by reading, not run on hardware
//
#include <hip/hip_runtime.h>

#define DI __device__ __forceinline__

typedef __attribute__((ext_vector_type(16))) __bf16        v16bf;
typedef __attribute__((ext_vector_type(8)))  float         v8f;
typedef __attribute__((ext_vector_type(4)))  unsigned int  u32x4;
typedef __attribute__((ext_vector_type(2)))  unsigned int  u32x2;
typedef __attribute__((ext_vector_type(4)))  float         f32x4;

union FragBf {
  u32x4        q[2];
  unsigned int u[8];
  v16bf        v;
};

DI unsigned short f2bf(float f) {
  unsigned int u = __builtin_bit_cast(unsigned int, f);
  unsigned int r = u + 0x7FFFu + ((u >> 16) & 1u);   // round-to-nearest-even
  return (unsigned short)(r >> 16);
}
DI unsigned int packbf(float lo, float hi) {
  return (unsigned int)f2bf(lo) | ((unsigned int)f2bf(hi) << 16);
}

// ---------------------------------------------------------------------------
// CDNA5 async global->LDS (16B per lane), tracked by ASYNCcnt.
// Builtin signature: (int4 addrspace(1)*, int4 addrspace(3)*, imm offset, imm cpol)
// ---------------------------------------------------------------------------
#if __has_builtin(__builtin_amdgcn_global_load_async_to_lds_b128) && \
    __has_builtin(__builtin_amdgcn_s_wait_asynccnt)
typedef int i32x4v __attribute__((__vector_size__(16)));
typedef __attribute__((address_space(1))) i32x4v* as1_i32x4p;
typedef __attribute__((address_space(3))) i32x4v* as3_i32x4p;
DI void async_cp16(void* lds, const void* g) {
  __builtin_amdgcn_global_load_async_to_lds_b128(
      (as1_i32x4p)(unsigned long long)g,
      (as3_i32x4p)(unsigned int)(unsigned long long)lds, 0, 0);
}
DI void async_wait0() { __builtin_amdgcn_s_wait_asynccnt(0); }
#else
DI void async_cp16(void* lds, const void* g) {
  *(u32x4*)lds = *(const u32x4*)g;
}
DI void async_wait0() {}
#endif

// ---------------------------------------------------------------------------
// fp32 -> bf16 elementwise (vectorized x4)
// ---------------------------------------------------------------------------
__global__ __launch_bounds__(256) void cvt_f32_bf16(const float* __restrict__ in,
                                                    unsigned short* __restrict__ out,
                                                    int n4) {
  int i = blockIdx.x * blockDim.x + threadIdx.x;
  if (i < n4) {
    f32x4 v = *(const f32x4*)(in + (size_t)i * 4);
    u32x2 p;
    p.x = packbf(v.x, v.y);
    p.y = packbf(v.z, v.w);
    *(u32x2*)(out + (size_t)i * 4) = p;
  }
}

// ---------------------------------------------------------------------------
// fp32 [K][N] -> bf16 [N][K] (transpose + convert, for weights)
// ---------------------------------------------------------------------------
__global__ __launch_bounds__(256) void cvt_transpose_bf16(const float* __restrict__ in,
                                                          unsigned short* __restrict__ out,
                                                          int K, int N) {
  int n = blockIdx.x * 32 + (int)threadIdx.x;
  int k = blockIdx.y * 8 + (int)threadIdx.y;
  if (n < N && k < K)
    out[(size_t)n * K + k] = f2bf(in[(size_t)k * N + n]);
}

// ---------------------------------------------------------------------------
// C[M][N] = A[M][K](bf16) * Bt[N][K](bf16) + bias[N];  OUT = f32 or bf16.
// Tile 128x128, BK=32, 8 waves (each 64x32 = 4x2 WMMA tiles).
// Double-buffered LDS, async staging, ONE barrier per k-step.
// ---------------------------------------------------------------------------
template <bool BF16_OUT>
__global__ __launch_bounds__(256) void gemm_bf16_wmma(
    const unsigned short* __restrict__ A,
    const unsigned short* __restrict__ Bt,
    const float* __restrict__ bias,
    void* __restrict__ Cout,
    int M, int N, int K) {
  __shared__ __align__(16) unsigned short As[2][128][40];
  __shared__ __align__(16) unsigned short Bs[2][128][40];

  const int tid  = threadIdx.x;
  const int wv   = tid >> 5;
  const int lane = tid & 31;
  const int m    = lane & 15;
  const int half = lane >> 4;
  const int Mb   = (wv & 1) * 64;
  const int Nb   = (wv >> 1) * 32;
  const int m0   = blockIdx.y * 128;
  const int n0   = blockIdx.x * 128;

  const int ldRow = tid >> 1;        // 0..127
  const int ldCol = (tid & 1) * 16;  // 0 or 16

  auto stage = [&](int buf, int kb) {
    const unsigned short* ag = A  + (size_t)(m0 + ldRow) * K + kb + ldCol;
    const unsigned short* bg = Bt + (size_t)(n0 + ldRow) * K + kb + ldCol;
    async_cp16(&As[buf][ldRow][ldCol],     ag);
    async_cp16(&As[buf][ldRow][ldCol + 8], ag + 8);
    async_cp16(&Bs[buf][ldRow][ldCol],     bg);
    async_cp16(&Bs[buf][ldRow][ldCol + 8], bg + 8);
  };

  const v8f zero = {0.f, 0.f, 0.f, 0.f, 0.f, 0.f, 0.f, 0.f};
  v8f acc[4][2];
#pragma unroll
  for (int i = 0; i < 4; ++i)
#pragma unroll
    for (int j = 0; j < 2; ++j) acc[i][j] = zero;

  stage(0, 0);
  const int nk = K >> 5;
  for (int i = 0; i < nk; ++i) {
    const int cur = i & 1;
    async_wait0();      // my staging into buf[cur] complete
    __syncthreads();    // everyone's staging complete; buf[cur^1] reads done
    if (i + 1 < nk) stage(cur ^ 1, (i + 1) * 32);

    FragBf af[4], bf[2];
#pragma unroll
    for (int mt = 0; mt < 4; ++mt) {
      int r = Mb + mt * 16 + m;
      af[mt].q[0] = *(const u32x4*)&As[cur][r][half * 8];
      af[mt].q[1] = *(const u32x4*)&As[cur][r][16 + half * 8];
    }
#pragma unroll
    for (int nt = 0; nt < 2; ++nt) {
      int r = Nb + nt * 16 + m;
      bf[nt].q[0] = *(const u32x4*)&Bs[cur][r][half * 8];
      bf[nt].q[1] = *(const u32x4*)&Bs[cur][r][16 + half * 8];
    }
#pragma unroll
    for (int mt = 0; mt < 4; ++mt)
#pragma unroll
      for (int nt = 0; nt < 2; ++nt)
        acc[mt][nt] = __builtin_amdgcn_wmma_f32_16x16x32_bf16(
            false, af[mt].v, false, bf[nt].v, (short)0, acc[mt][nt], false, false);
  }

#pragma unroll
  for (int nt = 0; nt < 2; ++nt) {
    int gcol = n0 + Nb + nt * 16 + m;
    float bv = bias[gcol];
#pragma unroll
    for (int mt = 0; mt < 4; ++mt) {
#pragma unroll
      for (int r = 0; r < 8; ++r) {
        int grow = m0 + Mb + mt * 16 + r + 8 * half;
        float val = acc[mt][nt][r] + bv;
        if (BF16_OUT)
          ((unsigned short*)Cout)[(size_t)grow * N + gcol] = f2bf(val);
        else
          ((float*)Cout)[(size_t)grow * N + gcol] = val;
      }
    }
  }
}

// ---------------------------------------------------------------------------
// Flash attention: qkv[B][T][3C] (bf16) -> Y[B][T][C] (bf16)
// block = (b, h, 128 q-rows); 8 waves, each owns 16 q-rows.
// Double-buffered K/V staging (async K, transpose-scatter V), one barrier/step.
// ---------------------------------------------------------------------------
__global__ __launch_bounds__(256) void flash_attn_wmma(
    const unsigned short* __restrict__ qkv, unsigned short* __restrict__ Y) {
  constexpr int T = 2048, Cd = 1024, HD = 64, ROW3 = 3072;

  __shared__ __align__(16) unsigned short Ks[2][32][72];    // [key][d] row-major
  __shared__ __align__(16) unsigned short Vts[2][64][40];   // [d][key] transposed
  __shared__ __align__(16) unsigned short Ps[8][16][40];    // per-wave P staging

  const int tid  = threadIdx.x;
  const int wv   = tid >> 5;
  const int lane = tid & 31;
  const int m    = lane & 15;
  const int half = lane >> 4;

  const int bid  = blockIdx.x;
  const int qblk = bid & 15;
  const int h    = (bid >> 4) & 15;
  const int b    = bid >> 8;
  const int q0   = qblk * 128;

  // --- Q fragments straight from bf16 memory ---
  FragBf qa[2];
  {
    const int tq = q0 + wv * 16 + m;
    const unsigned short* qrow = qkv + (size_t)(b * T + tq) * ROW3 + h * HD;
#pragma unroll
    for (int f = 0; f < 2; ++f)
#pragma unroll
      for (int v = 0; v < 8; ++v) {
        int d = f * 32 + (v >> 2) * 16 + half * 8 + (v & 3) * 2;
        qa[f].u[v] = *(const unsigned int*)(qrow + d);
      }
  }

  const int ldKey = tid >> 3;        // 0..31
  const int ldD   = (tid & 7) * 8;   // 0..56

  auto stage = [&](int buf, int k0) {
    const unsigned short* kr = qkv + (size_t)(b * T + k0 + ldKey) * ROW3 + Cd     + h * HD + ldD;
    const unsigned short* vr = qkv + (size_t)(b * T + k0 + ldKey) * ROW3 + 2 * Cd + h * HD + ldD;
    async_cp16(&Ks[buf][ldKey][ldD], kr);          // K rows: layout matches B-fragments
    u32x4 vv = *(const u32x4*)vr;                  // V: transpose-scatter bf16
    Vts[buf][ldD + 0][ldKey] = (unsigned short)(vv.x);
    Vts[buf][ldD + 1][ldKey] = (unsigned short)(vv.x >> 16);
    Vts[buf][ldD + 2][ldKey] = (unsigned short)(vv.y);
    Vts[buf][ldD + 3][ldKey] = (unsigned short)(vv.y >> 16);
    Vts[buf][ldD + 4][ldKey] = (unsigned short)(vv.z);
    Vts[buf][ldD + 5][ldKey] = (unsigned short)(vv.z >> 16);
    Vts[buf][ldD + 6][ldKey] = (unsigned short)(vv.w);
    Vts[buf][ldD + 7][ldKey] = (unsigned short)(vv.w >> 16);
  };

  const v8f zero = {0.f, 0.f, 0.f, 0.f, 0.f, 0.f, 0.f, 0.f};
  v8f O[4];
#pragma unroll
  for (int i = 0; i < 4; ++i) O[i] = zero;
  float mrow[8], lrow[8];
#pragma unroll
  for (int r = 0; r < 8; ++r) { mrow[r] = -3.0e38f; lrow[r] = 0.f; }

  const int nkb = q0 / 32 + 4;       // causal: keys [0, q0+128)
  stage(0, 0);

  for (int kb = 0; kb < nkb; ++kb) {
    const int cur = kb & 1;
    const int k0  = kb * 32;
    async_wait0();
    __syncthreads();
    if (kb + 1 < nkb) stage(cur ^ 1, k0 + 32);

    // --- S = Q * K^T : 16 x 32 keys ---
    v8f S0 = zero, S1 = zero;
#pragma unroll
    for (int s = 0; s < 2; ++s) {
      FragBf kf0, kf1;
      kf0.q[0] = *(const u32x4*)&Ks[cur][m][s * 32 + half * 8];
      kf0.q[1] = *(const u32x4*)&Ks[cur][m][s * 32 + 16 + half * 8];
      kf1.q[0] = *(const u32x4*)&Ks[cur][16 + m][s * 32 + half * 8];
      kf1.q[1] = *(const u32x4*)&Ks[cur][16 + m][s * 32 + 16 + half * 8];
      S0 = __builtin_amdgcn_wmma_f32_16x16x32_bf16(false, qa[s].v, false, kf0.v,
                                                   (short)0, S0, false, false);
      S1 = __builtin_amdgcn_wmma_f32_16x16x32_bf16(false, qa[s].v, false, kf1.v,
                                                   (short)0, S1, false, false);
    }

    // --- scale + causal mask + online softmax ---
    float corr[8];
#pragma unroll
    for (int r = 0; r < 8; ++r) {
      int qi = q0 + wv * 16 + r + 8 * half;
      float s0 = (k0 + m      <= qi) ? S0[r] * 0.125f : -3.0e38f;
      float s1 = (k0 + 16 + m <= qi) ? S1[r] * 0.125f : -3.0e38f;
      float mx = fmaxf(s0, s1);
      mx = fmaxf(mx, __shfl_xor(mx, 1));
      mx = fmaxf(mx, __shfl_xor(mx, 2));
      mx = fmaxf(mx, __shfl_xor(mx, 4));
      mx = fmaxf(mx, __shfl_xor(mx, 8));
      float nm = fmaxf(mrow[r], mx);
      corr[r] = __expf(mrow[r] - nm);
      mrow[r] = nm;
      float p0 = __expf(s0 - nm);
      float p1 = __expf(s1 - nm);
      S0[r] = p0; S1[r] = p1;
      float rs = p0 + p1;
      rs += __shfl_xor(rs, 1);
      rs += __shfl_xor(rs, 2);
      rs += __shfl_xor(rs, 4);
      rs += __shfl_xor(rs, 8);
      lrow[r] = lrow[r] * corr[r] + rs;
    }
#pragma unroll
    for (int dn = 0; dn < 4; ++dn)
#pragma unroll
      for (int r = 0; r < 8; ++r) O[dn][r] *= corr[r];

    // --- P: C-layout -> A-layout via wave-private LDS ---
#pragma unroll
    for (int r = 0; r < 8; ++r) {
      int rr = r + 8 * half;
      Ps[wv][rr][m]      = f2bf(S0[r]);
      Ps[wv][rr][16 + m] = f2bf(S1[r]);
    }
    FragBf pa;
    pa.q[0] = *(const u32x4*)&Ps[wv][m][half * 8];
    pa.q[1] = *(const u32x4*)&Ps[wv][m][16 + half * 8];

    // --- O += P * V ---
#pragma unroll
    for (int dn = 0; dn < 4; ++dn) {
      FragBf vf;
      vf.q[0] = *(const u32x4*)&Vts[cur][dn * 16 + m][half * 8];
      vf.q[1] = *(const u32x4*)&Vts[cur][dn * 16 + m][16 + half * 8];
      O[dn] = __builtin_amdgcn_wmma_f32_16x16x32_bf16(false, pa.v, false, vf.v,
                                                      (short)0, O[dn], false, false);
    }
  }

  // --- normalize + write bf16 ---
#pragma unroll
  for (int r = 0; r < 8; ++r) lrow[r] = 1.0f / lrow[r];
  const int tq = q0 + wv * 16;
#pragma unroll
  for (int dn = 0; dn < 4; ++dn) {
    int d = dn * 16 + m;
#pragma unroll
    for (int r = 0; r < 8; ++r) {
      int row = tq + r + 8 * half;
      Y[(size_t)(b * T + row) * Cd + h * HD + d] = f2bf(O[dn][r] * lrow[r]);
    }
  }
}

// ---------------------------------------------------------------------------
extern "C" void kernel_launch(void* const* d_in, const int* in_sizes, int n_in,
                              void* d_out, int out_size, void* d_ws, size_t ws_size,
                              hipStream_t stream) {
  (void)in_sizes; (void)n_in; (void)out_size; (void)ws_size;
  const float* x      = (const float*)d_in[0];
  const float* w_attn = (const float*)d_in[1];
  const float* b_attn = (const float*)d_in[2];
  const float* w_proj = (const float*)d_in[3];
  const float* b_proj = (const float*)d_in[4];
  float* out = (float*)d_out;

  // workspace layout (16B-aligned)
  char* ws = (char*)d_ws;
  unsigned short* xb   = (unsigned short*)(ws);              // 8192*1024 bf16 = 16 MiB
  unsigned short* wat  = (unsigned short*)(ws + 16777216);   // 3072*1024 bf16 =  6 MiB (transposed)
  unsigned short* wpt  = (unsigned short*)(ws + 23068672);   // 1024*1024 bf16 =  2 MiB (transposed)
  unsigned short* qkvb = (unsigned short*)(ws + 25165824);   // 8192*3072 bf16 = 48 MiB
  unsigned short* yb   = (unsigned short*)(ws + 75497472);   // 8192*1024 bf16 = 16 MiB

  cvt_f32_bf16<<<(8192 * 1024 / 4 + 255) / 256, 256, 0, stream>>>(x, xb, 8192 * 1024 / 4);
  cvt_transpose_bf16<<<dim3(3072 / 32, 1024 / 8), dim3(32, 8), 0, stream>>>(w_attn, wat, 1024, 3072);
  cvt_transpose_bf16<<<dim3(1024 / 32, 1024 / 8), dim3(32, 8), 0, stream>>>(w_proj, wpt, 1024, 1024);

  // qkv (bf16) = x @ w_attn + b_attn   (M=8192, N=3072, K=1024)
  gemm_bf16_wmma<true><<<dim3(3072 / 128, 8192 / 128), 256, 0, stream>>>(
      xb, wat, b_attn, qkvb, 8192, 3072, 1024);

  // flash attention -> yb (bf16)
  flash_attn_wmma<<<dim3(4 * 16 * 16), 256, 0, stream>>>(qkvb, yb);

  // out (f32) = y @ w_proj + b_proj    (M=8192, N=1024, K=1024)
  gemm_bf16_wmma<false><<<dim3(1024 / 128, 8192 / 128), 256, 0, stream>>>(
      yb, wpt, b_proj, out, 8192, 1024, 1024);
}
